// SpatiotemporalAttention_19344532701851
// MI455X (gfx1250) — matrix operand order repaired and checked by reading, not compiled, not run on hardware
//
#include <hip/hip_runtime.h>
#include <hip/hip_bf16.h>
#include <math.h>

#define BB   4
#define CC   64
#define CIc  32
#define NNc  4096
#define EPSc 1e-5f

typedef __attribute__((ext_vector_type(16))) __bf16 v16bf;
typedef __attribute__((ext_vector_type(8)))  float  v8f;

union FragB { uint4 q[2]; v16bf v; };

__device__ __forceinline__ unsigned short f2bf(float f) {
  unsigned int u = __float_as_uint(f);
  unsigned int r = (u + 0x7FFFu + ((u >> 16) & 1u)) >> 16;  // RNE
  return (unsigned short)r;
}
__device__ __forceinline__ float bf2f(unsigned short s) {
  return __uint_as_float(((unsigned int)s) << 16);
}

// ---------------------------------------------------------------------------
// K0: fold BN into 1x1 conv weights.
//  proj (BN->conv):  W'[o,c] = W[o,c]*sc[c];  b'[o] = b[o] + sum_c W[o,c]*(beta-mean*sc)
//  out  (conv->BN):  W''[o,ci] = sw[o]*W[o,ci]; b''[o] = sw[o]*(b[o]-mean[o]) + beta[o]
// ---------------------------------------------------------------------------
__global__ void fold_params(
    const float* gw, const float* gb, const float* gga, const float* gbe, const float* gmu, const float* gva,
    const float* tw, const float* tb, const float* tga, const float* tbe, const float* tmu, const float* tva,
    const float* pw, const float* pb, const float* pga, const float* pbe, const float* pmu, const float* pva,
    const float* ww, const float* wb, const float* wga, const float* wbe, const float* wmu, const float* wva,
    float* Wg, float* bg, float* Wt, float* bt, float* Wp, float* bp, float* Ww, float* bw)
{
  const int t = threadIdx.x;  // 64 threads
  if (t < CIc) {
    float s = 0.f;
    for (int c = 0; c < CC; ++c) {
      float sc = gga[c] * rsqrtf(gva[c] + EPSc);
      float of = gbe[c] - gmu[c] * sc;
      float w  = gw[t * CC + c];
      Wg[t * CC + c] = w * sc;  s += w * of;
    }
    bg[t] = gb[t] + s;
    s = 0.f;
    for (int c = 0; c < CC; ++c) {
      float sc = tga[c] * rsqrtf(tva[c] + EPSc);
      float of = tbe[c] - tmu[c] * sc;
      float w  = tw[t * CC + c];
      Wt[t * CC + c] = w * sc;  s += w * of;
    }
    bt[t] = tb[t] + s;
    s = 0.f;
    for (int c = 0; c < CC; ++c) {
      float sc = pga[c] * rsqrtf(pva[c] + EPSc);
      float of = pbe[c] - pmu[c] * sc;
      float w  = pw[t * CC + c];
      Wp[t * CC + c] = w * sc;  s += w * of;
    }
    bp[t] = pb[t] + s;
  }
  if (t < CC) {
    float sc = wga[t] * rsqrtf(wva[t] + EPSc);
    for (int ci = 0; ci < CIc; ++ci) Ww[t * CIc + ci] = ww[t * CIc + ci] * sc;
    bw[t] = sc * (wb[t] - wmu[t]) + wbe[t];
  }
}

// ---------------------------------------------------------------------------
// K1: projection GEMM (folded BN+conv), bf16 outputs in WMMA-friendly layouts.
//  outA: [B][CI][N] row-major  (WMMA-A for the value matmul)
//  outBT:[B][N][CI]            (serves as WMMA-A rows AND WMMA-B cols, K=c contiguous)
// ---------------------------------------------------------------------------
__global__ void proj_bn_conv(const float* __restrict__ x,
                             const float* __restrict__ Wa, const float* __restrict__ ba,
                             const float* __restrict__ Wb, const float* __restrict__ bb,
                             unsigned short* __restrict__ outA,
                             unsigned short* __restrict__ outBT)
{
  const int b = blockIdx.z;
  const int n = blockIdx.x * blockDim.x + threadIdx.x;
  const float* xb = x + (size_t)b * CC * NNc;
  float accA[CIc], accB[CIc];
#pragma unroll
  for (int o = 0; o < CIc; ++o) { accA[o] = 0.f; accB[o] = 0.f; }
  for (int c = 0; c < CC; ++c) {
    float xv = xb[c * NNc + n];
#pragma unroll
    for (int o = 0; o < CIc; ++o) {
      accA[o] = fmaf(Wa[o * CC + c], xv, accA[o]);   // uniform index -> s_load
      accB[o] = fmaf(Wb[o * CC + c], xv, accB[o]);
    }
  }
  unsigned short* oA = outA + (size_t)b * CIc * NNc;
  unsigned short* oT = outBT + ((size_t)b * NNc + n) * CIc;
#pragma unroll
  for (int o = 0; o < CIc; ++o) {
    oA[o * NNc + n] = f2bf(accA[o] + ba[o]);
    oT[o]           = f2bf(accB[o] + bb[o]);
  }
}

// ---------------------------------------------------------------------------
// K2: column softmax stats. E[n,m] = sum_c thT[n,c]*phT[m,c].
// One wave owns 16 m-columns. Lane-LOCAL online max/sum over this lane's 8
// rows per tile (no cross-lane traffic in the loop); lanes l and l^16 are
// merged exactly once at the end.
// ---------------------------------------------------------------------------
__global__ void __launch_bounds__(128) attn_stats(
    const unsigned short* __restrict__ thT, const unsigned short* __restrict__ phT,
    float* __restrict__ colmax, float* __restrict__ colsum)
{
  const int b    = blockIdx.y;
  const int wave = threadIdx.x >> 5;
  const int lane = threadIdx.x & 31;
  const int hi   = lane >> 4;
  const int lm   = lane & 15;
  const int m0   = (blockIdx.x * 4 + wave) * 16;
  const unsigned short* thTb = thT + (size_t)b * NNc * CIc;
  const unsigned short* phTb = phT + (size_t)b * NNc * CIc;

  FragB qf;  // B fragment: col m = m0+lm, K = c (contiguous in phT row)
  {
    const unsigned short* p = phTb + (size_t)(m0 + lm) * CIc + hi * 8;
    qf.q[0] = *(const uint4*)(p);
    qf.q[1] = *(const uint4*)(p + 16);
  }
  float runmax = -3.0e38f, runsum = 0.f;
  for (int it = 0; it < NNc / 16; ++it) {
    FragB af;  // A fragment: row n = it*16+lm, K = c
    const unsigned short* p = thTb + (size_t)(it * 16 + lm) * CIc + hi * 8;
    af.q[0] = *(const uint4*)(p);
    af.q[1] = *(const uint4*)(p + 16);
    v8f d = {0.f, 0.f, 0.f, 0.f, 0.f, 0.f, 0.f, 0.f};
    d = __builtin_amdgcn_wmma_f32_16x16x32_bf16(false, af.v, false, qf.v,
                                                (short)0, d, false, false);
    float tmax = d[0];
#pragma unroll
    for (int i = 1; i < 8; ++i) tmax = fmaxf(tmax, d[i]);
    float newmax = fmaxf(runmax, tmax);
    float s = 0.f;
#pragma unroll
    for (int i = 0; i < 8; ++i) s += __expf(d[i] - newmax);
    runsum = runsum * __expf(runmax - newmax) + s;
    runmax = newmax;
  }
  // merge lane pair (l, l^16): column m = m0+lm owned by both
  float pm = __shfl_xor(runmax, 16, 32);
  float ps = __shfl_xor(runsum, 16, 32);
  float m  = fmaxf(runmax, pm);
  float s  = runsum * __expf(runmax - m) + ps * __expf(pm - m);
  if (lane < 16) {
    colmax[b * NNc + m0 + lane] = m;
    colsum[b * NNc + m0 + lane] = s;
  }
}

// K3: per-batch global max of colmax (keeps pass-C exponents <= 0)
__global__ void gmax_reduce(const float* __restrict__ colmax, float* __restrict__ gmax)
{
  __shared__ float sm[256];
  const int b = blockIdx.x;
  float m = -3.0e38f;
  for (int i = threadIdx.x; i < NNc; i += 256) m = fmaxf(m, colmax[b * NNc + i]);
  sm[threadIdx.x] = m;
  __syncthreads();
  for (int s = 128; s > 0; s >>= 1) {
    if (threadIdx.x < s) sm[threadIdx.x] = fmaxf(sm[threadIdx.x], sm[threadIdx.x + s]);
    __syncthreads();
  }
  if (threadIdx.x == 0) gmax[b] = sm[0];
}

// K4: fold softmax normalizer into g2 columns: g2s[c,m] = g2[c,m]*exp(M*-mx[m])/Z[m]
__global__ void g2_scale(const unsigned short* __restrict__ g2,
                         const float* __restrict__ colmax, const float* __restrict__ colsum,
                         const float* __restrict__ gmax, unsigned short* __restrict__ g2s)
{
  const int idx = blockIdx.x * blockDim.x + threadIdx.x;  // B*CI*N
  const int b = idx / (CIc * NNc);
  const int m = idx & (NNc - 1);
  float sc = __expf(gmax[b] - colmax[b * NNc + m]) / colsum[b * NNc + m];
  g2s[idx] = f2bf(bf2f(g2[idx]) * sc);
}

// ---------------------------------------------------------------------------
// K5: flash attention-apply; y1 and y2 passes fused in one launch (blockIdx.z):
//  z=0: Aq=thT, Bq=phT, V=g1,  bias=colmax[col], rz=1/colsum[col]  -> y1
//  z=1: Aq=phT, Bq=thT, V=g2s, bias=gmax[b],     rz=1              -> y2
// One wave owns 16 output columns; loop over K in steps of 32, with
// global_prefetch of the next k-tile's A and V rows.
// ---------------------------------------------------------------------------
__global__ void __launch_bounds__(128) attn_apply(
    const unsigned short* __restrict__ thT,   // [B][N][CI]
    const unsigned short* __restrict__ phT,   // [B][N][CI]
    const unsigned short* __restrict__ g1,    // [B][CI][N]
    const unsigned short* __restrict__ g2s,   // [B][CI][N]
    const float* __restrict__ colmax, const float* __restrict__ colsum,
    const float* __restrict__ gmax,
    float* __restrict__ y1, float* __restrict__ y2)   // [B][CI][N]
{
  __shared__ __align__(16) unsigned short Pl[4][16 * 40];  // per-wave P tile, ld=40 (pad)
  const int mode = blockIdx.z;
  const int b    = blockIdx.y;
  const int wave = threadIdx.x >> 5;
  const int lane = threadIdx.x & 31;
  const int hi   = lane >> 4;
  const int lm   = lane & 15;
  const int col0 = (blockIdx.x * 4 + wave) * 16;
  const unsigned short* Ab = (mode == 0 ? thT : phT) + (size_t)b * NNc * CIc;
  const unsigned short* Bb = (mode == 0 ? phT : thT) + (size_t)b * NNc * CIc;
  const unsigned short* Vb = (mode == 0 ? g1 : g2s) + (size_t)b * CIc * NNc;
  float* Y = (mode == 0 ? y1 : y2);
  unsigned short* P = &Pl[wave][0];

  FragB bq;  // query-side B fragment: col = col0+lm, K = c  (constant over the K loop)
  {
    const unsigned short* p = Bb + (size_t)(col0 + lm) * CIc + hi * 8;
    bq.q[0] = *(const uint4*)(p);
    bq.q[1] = *(const uint4*)(p + 16);
  }

  float cm, rz;
  if (mode == 0) {
    cm = colmax[b * NNc + col0 + lm];
    rz = 1.0f / colsum[b * NNc + col0 + lm];
  } else {
    cm = gmax[b];
    rz = 1.0f;
  }

  v8f acc0 = {0.f, 0.f, 0.f, 0.f, 0.f, 0.f, 0.f, 0.f};
  v8f acc1 = {0.f, 0.f, 0.f, 0.f, 0.f, 0.f, 0.f, 0.f};

  for (int k0 = 0; k0 < NNc; k0 += 32) {
    // Prefetch next k-tile rows into near caches (global_prefetch_b8);
    // addresses stay inside the workspace allocation, OOB tails are benign.
    __builtin_prefetch(Ab + (size_t)(k0 + 32 + lm) * CIc, 0, 3);
    __builtin_prefetch(Ab + (size_t)(k0 + 48 + lm) * CIc, 0, 3);
    __builtin_prefetch(Vb + (size_t)lm * NNc + k0 + 32, 0, 3);
    __builtin_prefetch(Vb + (size_t)(16 + lm) * NNc + k0 + 32, 0, 3);
#pragma unroll
    for (int fi = 0; fi < 2; ++fi) {
      FragB af;  // rows k0+fi*16 .. +15 of AqT
      const unsigned short* p = Ab + (size_t)(k0 + fi * 16 + lm) * CIc + hi * 8;
      af.q[0] = *(const uint4*)(p);
      af.q[1] = *(const uint4*)(p + 16);
      v8f d = {0.f, 0.f, 0.f, 0.f, 0.f, 0.f, 0.f, 0.f};
      d = __builtin_amdgcn_wmma_f32_16x16x32_bf16(false, af.v, false, bq.v,
                                                  (short)0, d, false, false);
      union { unsigned short s[8]; uint4 q; } pk;
#pragma unroll
      for (int i = 0; i < 8; ++i) pk.s[i] = f2bf(__expf(d[i] - cm));
      // store P^T: LDS[col_local][k_local], k contiguous (B-fragment layout)
      *(uint4*)(P + lm * 40 + fi * 16 + hi * 8) = pk.q;
    }
    // same-wave DS ops are in-order: no barrier needed (per-wave private LDS tile)
    FragB b2;  // P as B fragment: col = lm, K = k_local contiguous
    {
      const unsigned short* p = P + lm * 40 + hi * 8;
      b2.q[0] = *(const uint4*)(p);
      b2.q[1] = *(const uint4*)(p + 16);
    }
#pragma unroll
    for (int ct = 0; ct < 2; ++ct) {
      FragB a2;  // V rows c = ct*16+lm, K = k0.. contiguous
      const unsigned short* p = Vb + (size_t)(ct * 16 + lm) * NNc + k0 + hi * 8;
      a2.q[0] = *(const uint4*)(p);
      a2.q[1] = *(const uint4*)(p + 16);
      if (ct == 0)
        acc0 = __builtin_amdgcn_wmma_f32_16x16x32_bf16(false, a2.v, false, b2.v,
                                                       (short)0, acc0, false, false);
      else
        acc1 = __builtin_amdgcn_wmma_f32_16x16x32_bf16(false, a2.v, false, b2.v,
                                                       (short)0, acc1, false, false);
    }
  }
  float* Yb = Y + (size_t)b * CIc * NNc;
#pragma unroll
  for (int i = 0; i < 8; ++i) {
    Yb[(hi * 8 + i) * NNc + col0 + lm]        = acc0[i] * rz;
    Yb[(16 + hi * 8 + i) * NNc + col0 + lm]   = acc1[i] * rz;
  }
}

// K6: out = x + W''@y + b''  (folded conv+BN + residual); both outputs in one
// launch via blockIdx.y.
__global__ void out_epilogue(const float* __restrict__ x1, const float* __restrict__ x2,
                             const float* __restrict__ y1, const float* __restrict__ y2,
                             const float* __restrict__ Ww, const float* __restrict__ bw,
                             float* __restrict__ out)
{
  const int sel = blockIdx.y;
  const int idx = blockIdx.x * blockDim.x + threadIdx.x;  // B*N
  const int b = idx / NNc;
  const int n = idx & (NNc - 1);
  const float* x   = sel ? x2 : x1;
  const float* Yin = sel ? y2 : y1;
  float* o = out + (size_t)sel * BB * CC * NNc;
  const float* yb = Yin + (size_t)b * CIc * NNc + n;
  float yv[CIc];
#pragma unroll
  for (int ci = 0; ci < CIc; ++ci) yv[ci] = yb[ci * NNc];
  const float* xb = x + (size_t)b * CC * NNc + n;
  float* ob = o + (size_t)b * CC * NNc + n;
  for (int c = 0; c < CC; ++c) {
    float a = bw[c];
#pragma unroll
    for (int ci = 0; ci < CIc; ++ci) a = fmaf(Ww[c * CIc + ci], yv[ci], a);
    ob[c * NNc] = xb[c * NNc] + a;
  }
}

extern "C" void kernel_launch(void* const* d_in, const int* in_sizes, int n_in,
                              void* d_out, int out_size, void* d_ws, size_t ws_size,
                              hipStream_t stream) {
  (void)in_sizes; (void)n_in; (void)out_size; (void)ws_size;
  const float* x1 = (const float*)d_in[0];
  const float* x2 = (const float*)d_in[1];
  const float* gp[6];  for (int i = 0; i < 6; ++i) gp[i] = (const float*)d_in[2 + i];
  const float* tp[6];  for (int i = 0; i < 6; ++i) tp[i] = (const float*)d_in[8 + i];
  const float* pp[6];  for (int i = 0; i < 6; ++i) pp[i] = (const float*)d_in[14 + i];
  const float* wp[6];  for (int i = 0; i < 6; ++i) wp[i] = (const float*)d_in[20 + i];

  char* ws = (char*)d_ws;
  size_t off = 0;
  auto alloc = [&](size_t bytes) -> void* {
    void* p = ws + off;
    off = (off + bytes + 255) & ~(size_t)255;
    return p;
  };
  float* Wg = (float*)alloc(CIc * CC * 4);  float* bg = (float*)alloc(CIc * 4);
  float* Wt = (float*)alloc(CIc * CC * 4);  float* bt = (float*)alloc(CIc * 4);
  float* Wp = (float*)alloc(CIc * CC * 4);  float* bp = (float*)alloc(CIc * 4);
  float* Ww = (float*)alloc(CC * CIc * 4);  float* bw = (float*)alloc(CC * 4);
  unsigned short* g1  = (unsigned short*)alloc((size_t)BB * CIc * NNc * 2);
  unsigned short* g2  = (unsigned short*)alloc((size_t)BB * CIc * NNc * 2);
  unsigned short* g2s = (unsigned short*)alloc((size_t)BB * CIc * NNc * 2);
  unsigned short* thT = (unsigned short*)alloc((size_t)BB * NNc * CIc * 2);
  unsigned short* phT = (unsigned short*)alloc((size_t)BB * NNc * CIc * 2);
  float* colmax = (float*)alloc((size_t)BB * NNc * 4);
  float* colsum = (float*)alloc((size_t)BB * NNc * 4);
  float* gmax   = (float*)alloc(BB * 4);
  float* y1 = (float*)alloc((size_t)BB * CIc * NNc * 4);
  float* y2 = (float*)alloc((size_t)BB * CIc * NNc * 4);

  fold_params<<<1, 64, 0, stream>>>(
      gp[0], gp[1], gp[2], gp[3], gp[4], gp[5],
      tp[0], tp[1], tp[2], tp[3], tp[4], tp[5],
      pp[0], pp[1], pp[2], pp[3], pp[4], pp[5],
      wp[0], wp[1], wp[2], wp[3], wp[4], wp[5],
      Wg, bg, Wt, bt, Wp, bp, Ww, bw);

  // x1 -> g1 ([CI][N]) and thT ([N][CI]); x2 -> g2 and phT
  proj_bn_conv<<<dim3(NNc / 256, 1, BB), 256, 0, stream>>>(x1, Wg, bg, Wt, bt, g1, thT);
  proj_bn_conv<<<dim3(NNc / 256, 1, BB), 256, 0, stream>>>(x2, Wg, bg, Wp, bp, g2, phT);

  attn_stats<<<dim3(NNc / 64, BB), 128, 0, stream>>>(thT, phT, colmax, colsum);
  gmax_reduce<<<BB, 256, 0, stream>>>(colmax, gmax);
  g2_scale<<<(BB * CIc * NNc) / 256, 256, 0, stream>>>(g2, colmax, colsum, gmax, g2s);

  // y1 (z=0) and y2 (z=1) in a single launch: 2048 resident waves
  attn_apply<<<dim3(NNc / 64, BB, 2), 128, 0, stream>>>(thT, phT, g1, g2s,
                                                        colmax, colsum, gmax, y1, y2);

  out_epilogue<<<dim3((BB * NNc) / 256, 2), 256, 0, stream>>>(x1, x2, y1, y2, Ww, bw,
                                                              (float*)d_out);
}
